// ELIAS_46231027974460
// MI455X (gfx1250) — compile-verified, matrix operand order checked by reading, not run on hardware
//
#include <hip/hip_runtime.h>

typedef __attribute__((ext_vector_type(16))) __bf16 v16bf;
typedef __attribute__((ext_vector_type(8)))  float  v8f;

#define D_DIM   768
#define B_SZ    256
#define NPAD    8256          // 8193 padded up to 129 * 64 (516 tiles)
#define NREAL   8193
#define NCLIP   8192          // out1 drops last column
#define BEAM    20
#define LEAF    64
#define S_TOT   (BEAM * LEAF) // 1280
#define ALPHA_F 10.0f

#define WMMA_BF16(a, b, c) \
    __builtin_amdgcn_wmma_f32_16x16x32_bf16(false, (a), false, (b), (short)0, (c), false, false)

// ---------------------------------------------------------------------------
// f32 -> bf16 conversion (zero-fills the padded tail region)
// ---------------------------------------------------------------------------
__global__ __launch_bounds__(256) void conv_bf16(const float* __restrict__ src,
                                                 __bf16* __restrict__ dst,
                                                 int n_src, int n_total) {
    int i = blockIdx.x * 256 + threadIdx.x;
    if (i < n_total) dst[i] = (i < n_src) ? (__bf16)src[i] : (__bf16)0.0f;
}

// ---------------------------------------------------------------------------
// WMMA GEMM: logits[256, NPAD] = embs_bf[256,768] @ w1_bf[NPAD,768]^T + bias
// Per-wave tile: 16(M) x 64(N), 4 independent accumulators share one A
// fragment per K-step.  Fragments are explicitly double-buffered: iteration
// k's WMMAs consume fragments loaded in iteration k-1, so the next 10 b128
// loads stay in flight behind the XDL pipe (partial s_wait_loadcnt).
// Grid: (256/128, NPAD/64) = (2, 129), 8 waves stacked along M.
// ---------------------------------------------------------------------------
__global__ __launch_bounds__(256) void wmma_gemm(const __bf16* __restrict__ A,
                                                 const __bf16* __restrict__ Bm,
                                                 const float* __restrict__ bias,
                                                 float* __restrict__ Cout) {
    const int wave = threadIdx.x >> 5;
    const int lane = threadIdx.x & 31;
    const int m_base = blockIdx.x * 128 + wave * 16;
    const int n_base = blockIdx.y * 64;

    const int lrow = lane & 15;
    const int koff = (lane >> 4) << 4;   // lanes 0-15: K 0..15, lanes 16-31: K 16..31
    const __bf16* ap = A  + (size_t)(m_base + lrow) * D_DIM + koff;
    const __bf16* bp = Bm + (size_t)(n_base + lrow) * D_DIM + koff;

    // prologue: fragments for k = 0
    v16bf a  = *(const v16bf*)(ap);
    v16bf b0 = *(const v16bf*)(bp);
    v16bf b1 = *(const v16bf*)(bp + 16 * D_DIM);
    v16bf b2 = *(const v16bf*)(bp + 32 * D_DIM);
    v16bf b3 = *(const v16bf*)(bp + 48 * D_DIM);

    v8f c0 = {}, c1 = {}, c2 = {}, c3 = {};
#pragma unroll 1
    for (int k = 32; k < D_DIM; k += 32) {
        // issue next iteration's loads before consuming current fragments
        v16bf an  = *(const v16bf*)(ap + k);
        v16bf bn0 = *(const v16bf*)(bp + k);
        v16bf bn1 = *(const v16bf*)(bp + 16 * D_DIM + k);
        v16bf bn2 = *(const v16bf*)(bp + 32 * D_DIM + k);
        v16bf bn3 = *(const v16bf*)(bp + 48 * D_DIM + k);
        c0 = WMMA_BF16(a, b0, c0);
        c1 = WMMA_BF16(a, b1, c1);
        c2 = WMMA_BF16(a, b2, c2);
        c3 = WMMA_BF16(a, b3, c3);
        a = an; b0 = bn0; b1 = bn1; b2 = bn2; b3 = bn3;
    }
    // epilogue: last K-step
    c0 = WMMA_BF16(a, b0, c0);
    c1 = WMMA_BF16(a, b1, c1);
    c2 = WMMA_BF16(a, b2, c2);
    c3 = WMMA_BF16(a, b3, c3);

    // D layout: vgpr r -> M = m_base + r + 8*(lane>=16), N = n_base + (lane&15)
    const int mrow = m_base + ((lane >> 4) << 3);
    v8f acc[4] = {c0, c1, c2, c3};
#pragma unroll
    for (int t = 0; t < 4; ++t) {
        const int nc = n_base + t * 16 + lrow;
        const float bn = (nc < NREAL) ? bias[nc] : -__builtin_inff();
        float* outp = Cout + (size_t)mrow * NPAD + nc;
#pragma unroll
        for (int r = 0; r < 8; ++r)
            outp[(size_t)r * NPAD] = acc[t][r] + bn;
    }
}

// ---------------------------------------------------------------------------
// Per-row softmax stats + iterative top-20 (ranking by logits == ranking by
// clipped softmax except for clip-ties, acceptable).  One block per row.
// ---------------------------------------------------------------------------
__global__ __launch_bounds__(256) void softmax_topk(float* __restrict__ logits,
                                                    float* __restrict__ topv,
                                                    int*   __restrict__ topi) {
    __shared__ float sred[256];
    __shared__ float smax[256];
    __shared__ int   sidx[256];
    const int b = blockIdx.x, t = threadIdx.x;
    float* row = logits + (size_t)b * NPAD;

    // row max (pad cols are -inf, harmless)
    float m = -__builtin_inff();
    for (int i = t; i < NPAD; i += 256) m = fmaxf(m, row[i]);
    sred[t] = m; __syncthreads();
    for (int s = 128; s > 0; s >>= 1) { if (t < s) sred[t] = fmaxf(sred[t], sred[t+s]); __syncthreads(); }
    m = sred[0]; __syncthreads();

    // sum of exp
    float z = 0.f;
    for (int i = t; i < NPAD; i += 256) z += __expf(row[i] - m);
    sred[t] = z; __syncthreads();
    for (int s = 128; s > 0; s >>= 1) { if (t < s) sred[t] += sred[t+s]; __syncthreads(); }
    z = sred[0]; __syncthreads();
    const float scale = ALPHA_F / z;

    for (int kk = 0; kk < BEAM; ++kk) {
        float bm = -__builtin_inff(); int bi = 0x7FFFFFFF;
        for (int i = t; i < NCLIP; i += 256) {
            float v = row[i];
            if (v > bm) { bm = v; bi = i; }
        }
        smax[t] = bm; sidx[t] = bi; __syncthreads();
        for (int s = 128; s > 0; s >>= 1) {
            if (t < s) {
                if (smax[t+s] > smax[t] ||
                    (smax[t+s] == smax[t] && sidx[t+s] < sidx[t])) {
                    smax[t] = smax[t+s]; sidx[t] = sidx[t+s];
                }
            }
            __syncthreads();
        }
        if (t == 0) {
            int idx = sidx[0];
            topi[b*BEAM + kk] = idx;
            topv[b*BEAM + kk] = fminf(__expf(smax[0] - m) * scale, 1.0f);
            row[idx] = -__builtin_inff();   // exclude; logits recomputed each call
        }
        __syncthreads();
    }
}

// ---------------------------------------------------------------------------
// Gathered leaf classifier: one wave per (b, shortlist slot).
// Bandwidth-bound phase: float4 loads + wave32 xor-shuffle reduction.
// ---------------------------------------------------------------------------
__global__ __launch_bounds__(256) void leaf_score(const float* __restrict__ embs,
                                                  const float* __restrict__ wleaf,
                                                  const float* __restrict__ bleaf,
                                                  const int*   __restrict__ Ai,
                                                  const float* __restrict__ Av,
                                                  const float* __restrict__ topv,
                                                  const int*   __restrict__ topi,
                                                  float* __restrict__ scores,
                                                  int*   __restrict__ short_out) {
    const int wave = threadIdx.x >> 5;
    const int lane = threadIdx.x & 31;
    const int g = blockIdx.x * 8 + wave;          // (b, j) flat index
    if (g >= B_SZ * S_TOT) return;
    const int b = g / S_TOT;
    const int j = g - b * S_TOT;
    const int kslot = j >> 6;                     // beam slot
    const int l     = j & 63;                     // leaf within cluster

    const int   cluster = topi[b*BEAM + kslot];
    const int   label   = Ai[(size_t)cluster * LEAF + l];
    const float sval    = Av[(size_t)cluster * LEAF + l] * topv[b*BEAM + kslot];

    const float4* e = (const float4*)(embs  + (size_t)b     * D_DIM);
    const float4* w = (const float4*)(wleaf + (size_t)label * D_DIM);
    float acc = 0.f;
#pragma unroll
    for (int i = 0; i < 6; ++i) {                 // 6 * 32 lanes * 4 = 768
        float4 ev = e[lane + 32*i];
        float4 wv = w[lane + 32*i];
        acc += ev.x*wv.x + ev.y*wv.y + ev.z*wv.z + ev.w*wv.w;
    }
#pragma unroll
    for (int off = 16; off > 0; off >>= 1)
        acc += __shfl_xor(acc, off, 32);

    if (lane == 0) {
        float out2 = acc + bleaf[label];
        scores[g]    = sval / (1.f + __expf(-out2));
        short_out[g] = label;                     // raw int32 in second half
    }
}

// ---------------------------------------------------------------------------
extern "C" void kernel_launch(void* const* d_in, const int* in_sizes, int n_in,
                              void* d_out, int out_size, void* d_ws, size_t ws_size,
                              hipStream_t stream) {
    const float* embs = (const float*)d_in[0];   // [256,768]
    const float* w1w  = (const float*)d_in[1];   // [8193,768]
    const float* w1b  = (const float*)d_in[2];   // [8193]
    const float* wl   = (const float*)d_in[3];   // [262145,768]
    const float* wlb  = (const float*)d_in[4];   // [262145]
    const int*   Ai   = (const int*)  d_in[5];   // [8192,64]
    const float* Av   = (const float*)d_in[6];   // [8192,64]

    // workspace layout (256B-aligned sections)
    char* ws = (char*)d_ws;
    const size_t o_embs_bf = 0;                                  // 256*768*2   = 393,216
    const size_t o_w1_bf   = o_embs_bf + 393216;                 // 8256*768*2  = 12,681,216
    const size_t o_logits  = o_w1_bf   + 12681216;               // 256*8256*4  = 8,454,144
    const size_t o_topv    = o_logits  + 8454144;                // 256*20*4
    const size_t o_topi    = o_topv    + 20480;                  // 256*20*4
    __bf16* embs_bf = (__bf16*)(ws + o_embs_bf);
    __bf16* w1_bf   = (__bf16*)(ws + o_w1_bf);
    float*  logits  = (float*) (ws + o_logits);
    float*  topv    = (float*) (ws + o_topv);
    int*    topi    = (int*)   (ws + o_topi);

    float* scores    = (float*)d_out;                 // [256*1280]
    int*   short_out = (int*)d_out + B_SZ * S_TOT;    // raw int32 second half

    const int n_w1  = NREAL * D_DIM;     // 6,292,224
    const int n_w1p = NPAD  * D_DIM;     // 6,340,608 (pad zero-filled)
    const int n_e   = B_SZ  * D_DIM;     // 196,608

    conv_bf16<<<(n_w1p + 255) / 256, 256, 0, stream>>>(w1w, w1_bf, n_w1, n_w1p);
    conv_bf16<<<(n_e   + 255) / 256, 256, 0, stream>>>(embs, embs_bf, n_e, n_e);

    dim3 ggrid(B_SZ / 128, NPAD / 64);                // (2, 129)
    wmma_gemm<<<ggrid, 256, 0, stream>>>(embs_bf, w1_bf, w1b, logits);

    softmax_topk<<<B_SZ, 256, 0, stream>>>(logits, topv, topi);

    const int total_waves = B_SZ * S_TOT;             // 327,680
    leaf_score<<<(total_waves + 7) / 8, 256, 0, stream>>>(
        embs, wl, wlb, Ai, Av, topv, topi, scores, short_out);
}
